// DeformableTransformer_28973849379505
// MI455X (gfx1250) — matrix operand order repaired and checked by reading, not compiled, not run on hardware
//
#include <hip/hip_runtime.h>
#include <hip/hip_bf16.h>

#define NUM_LAYERS 6
#define D_MODEL 256
#define N_HEADS 8
#define HEAD_DIM 32
#define D_FFN 1024
#define BZ 2
#define NQ_ 1600
#define S_TOTAL_ 5440

typedef __attribute__((ext_vector_type(16))) _Float16     v16h;
typedef __attribute__((ext_vector_type(8)))  float        v8f;
typedef __attribute__((ext_vector_type(4)))  unsigned int v4u;
typedef __attribute__((ext_vector_type(4)))  int          v4i;
typedef __attribute__((ext_vector_type(8)))  int          v8i;

typedef __attribute__((address_space(1))) v4i g_v4i;  // global
typedef __attribute__((address_space(3))) v4i l_v4i;  // LDS

#if __has_builtin(__builtin_amdgcn_global_load_async_to_lds_b128)
#define HAVE_ASYNC_LDS 1
#endif
#if __has_builtin(__builtin_amdgcn_tensor_load_to_lds)
#define HAVE_TDM 1
#endif

// A-operand (16-bit, 16x32) K index pattern from ISA table:
// lanes 0-15: K = e (e<8) or e+8 (e>=8); lanes 16-31: +8
__device__ __forceinline__ int kpatA(int e, int hi) {
    return e + (e < 8 ? 0 : 8) + (hi ? 8 : 0);
}

__device__ __forceinline__ void wait_async0() {
#if __has_builtin(__builtin_amdgcn_s_wait_asynccnt)
    __builtin_amdgcn_s_wait_asynccnt(0);
#else
    asm volatile("s_wait_asynccnt 0x0" ::: "memory");
#endif
}

__device__ __forceinline__ void wait_tensor0() {
#if __has_builtin(__builtin_amdgcn_s_wait_tensorcnt)
    __builtin_amdgcn_s_wait_tensorcnt(0);
#else
    asm volatile("s_wait_tensorcnt 0x0" ::: "memory");
#endif
}

#if defined(HAVE_ASYNC_LDS)
// Copy 16 bytes global -> LDS through the async (ASYNCcnt) path
__device__ __forceinline__ void async_cp16B(const float* g, float* l) {
    __builtin_amdgcn_global_load_async_to_lds_b128(
        (g_v4i*)(void*)g,
        (l_v4i*)(void*)l,
        0, 0);
}
#endif

#if defined(HAVE_TDM)
// TDM: load a 2D tile (rows x 32 f32 elements, global row stride `gstride`
// elements) into LDS with 1-DWORD padding after every 32 DWORDs
// (-> LDS row stride 33 floats). D# packed per CDNA5 ISA 8.3/8.4.
__device__ __forceinline__ void tdm_load_2d_32(const float* gsrc, void* lds_dst,
                                               int rows, int gstride) {
    unsigned long long lds = (unsigned)(size_t)lds_dst;
    unsigned long long ga  = (unsigned long long)(size_t)gsrc;

    // group0: count=1 | lds_addr<<32 ; global_addr | type=2
    unsigned long long g0q0 = 1ull | (lds << 32);
    unsigned long long g0q1 = ga | (2ull << 62);
    v4u g0 = { (unsigned)g0q0, (unsigned)(g0q0 >> 32),
               (unsigned)g0q1, (unsigned)(g0q1 >> 32) };

    // group1: data_size=2 (4B) | pad_enable | pad_interval=4 (32 DW) |
    //         pad_amount=0 (1 DW) | tensor_dim0=32 | tensor_dim1=rows |
    //         tile_dim0=32 | tile_dim1=rows | tensor_dim0_stride=gstride
    unsigned long long q0 = (2ull << 16) | (1ull << 20) | (4ull << 22) |
                            (32ull << 48);                       // tensor_dim0 lo16
    unsigned long long q1 = 0ull |                               // tensor_dim0 hi16
                            ((unsigned long long)(unsigned)rows << 16) |
                            (32ull << 48);                       // tile_dim0
    unsigned long long q2 = (unsigned long long)(unsigned)rows | // tile_dim1
                            ((unsigned long long)(unsigned)gstride << 32);
    unsigned long long q3 = 0ull;                                // dim1 stride unused (2D)
    v8i g1 = { (int)q0, (int)(q0 >> 32), (int)q1, (int)(q1 >> 32),
               (int)q2, (int)(q2 >> 32), (int)q3, (int)(q3 >> 32) };
    v4i z4 = {0, 0, 0, 0};
#if __clang_major__ >= 23
    v8i z8 = {0, 0, 0, 0, 0, 0, 0, 0};
    __builtin_amdgcn_tensor_load_to_lds(g0, g1, z4, z4, z8, 0);
#else
    __builtin_amdgcn_tensor_load_to_lds(g0, g1, z4, z4, 0);
#endif
}
#endif

// ---------------------------------------------------------------------------
// Tiled GEMM: C[M,N] = op(A[M,K] @ W[N,K]^T + bias)   (x @ w.T + b)
// 128-thread block -> 64x64 C tile; wave w -> rows w*16..w*16+15, 4 sub-tiles
// across N. A/W K-chunks staged in LDS (async path when available).
// mode: 0 = bias, 1 = bias+relu
// ---------------------------------------------------------------------------
__global__ void gemm_wmma_kernel(const float* __restrict__ A,
                                 const float* __restrict__ W,
                                 const float* __restrict__ bias,
                                 float* __restrict__ C,
                                 int M, int N, int K, int mode)
{
    __shared__ float As[64][33];
    __shared__ float Ws[64][33];

    const int lane = threadIdx.x & 31;
    const int wave = threadIdx.x >> 5;
    const int m0 = blockIdx.x * 64;
    const int n0 = blockIdx.y * 64;
    const int nl = lane & 15, hi = lane >> 4;

    v8f acc[4] = {{}, {}, {}, {}};

    // staging role: threads 0-63 stage A rows, threads 64-127 stage W rows
    const int  srow = threadIdx.x & 63;
    const bool isW  = threadIdx.x >= 64;
    const float* grow = isW ? (W + (long)(n0 + srow) * K)
                            : (A + (long)(m0 + srow) * K);
    float* lrow = isW ? Ws[srow] : As[srow];

    for (int k0 = 0; k0 < K; k0 += 32) {
#if defined(HAVE_ASYNC_LDS)
#pragma unroll
        for (int j = 0; j < 8; ++j)
            async_cp16B(grow + k0 + j * 4, lrow + j * 4);
        wait_async0();
#else
#pragma unroll
        for (int j = 0; j < 32; ++j) lrow[j] = grow[k0 + j];
#endif
        __syncthreads();

        v16h a;
#pragma unroll
        for (int e = 0; e < 16; ++e)
            a[e] = (_Float16)As[wave * 16 + nl][hi * 16 + e];
#pragma unroll
        for (int sub = 0; sub < 4; ++sub) {
            v16h b;
#pragma unroll
            for (int e = 0; e < 16; ++e)
                b[e] = (_Float16)Ws[sub * 16 + nl][hi * 16 + e];
            acc[sub] = __builtin_amdgcn_wmma_f32_16x16x32_f16(
                false, a, false, b, (short)0, acc[sub], false, false);
        }
        __syncthreads();
    }

#pragma unroll
    for (int sub = 0; sub < 4; ++sub) {
        int col = n0 + sub * 16 + nl;
        float bv = bias ? bias[col] : 0.f;
#pragma unroll
        for (int r = 0; r < 8; ++r) {
            int m = m0 + wave * 16 + r + hi * 8;
            float v = acc[sub][r] + bv;
            if (mode == 1) v = fmaxf(v, 0.f);
            C[(long)m * N + col] = v;
        }
    }
}

// ---------------------------------------------------------------------------
// Flash self-attention: 8 heads, head_dim 32, 16-query tile per wave.
// Q/K/V layouts: [b][token][h*32 + d]. K/V chunks staged via TDM when present.
// ---------------------------------------------------------------------------
#define FA_WPB 4
__global__ void flash_attn_kernel(const float* __restrict__ Qp,
                                  const float* __restrict__ Kp,
                                  const float* __restrict__ Vp,
                                  float* __restrict__ O, int nkv)
{
    __shared__ float    Ksf[32][33];
    __shared__ float    Vsf[32][33];
    __shared__ _Float16 Pt[FA_WPB][16][33];

    const int lane = threadIdx.x & 31;
    const int wave = threadIdx.x >> 5;
    const int t  = blockIdx.x * FA_WPB + wave;
    const int qt = t % (NQ_ / 16);
    const int h  = (t / (NQ_ / 16)) % N_HEADS;
    const int b  = t / ((NQ_ / 16) * N_HEADS);
    const int q0 = qt * 16;
    const int nl = lane & 15, hi = lane >> 4;
    const float scale = 0.17677669529663687f; // 1/sqrt(32)

    // Q A-operand (rows = queries, K-dim = 32 head dims)
    v16h qa;
    const float* qrow = Qp + (long)(b * NQ_ + q0 + nl) * D_MODEL + h * HEAD_DIM;
#pragma unroll
    for (int e = 0; e < 16; ++e)
        qa[e] = (_Float16)(qrow[hi * 16 + e] * scale);

    v8f acc0 = {}, acc1 = {};
    float m_run[8], l_run[8];
#pragma unroll
    for (int r = 0; r < 8; ++r) { m_run[r] = -3.0e38f; l_run[r] = 0.f; }

    const float* kbase = Kp + ((long)b * NQ_) * D_MODEL + h * HEAD_DIM;
    const float* vbase = Vp + ((long)b * NQ_) * D_MODEL + h * HEAD_DIM;

    for (int k0 = 0; k0 < nkv; k0 += 32) {
#if defined(HAVE_TDM)
        if (wave == 0) {
            tdm_load_2d_32(kbase + (long)k0 * D_MODEL, &Ksf[0][0], 32, D_MODEL);
            tdm_load_2d_32(vbase + (long)k0 * D_MODEL, &Vsf[0][0], 32, D_MODEL);
            wait_tensor0();
        }
#else
        {
            int row = threadIdx.x >> 1;
            int hsel = (threadIdx.x & 1) * 16;
            const float* sb = (row < 32) ? kbase : vbase;
            const float* s = sb + (long)(k0 + (row & 31)) * D_MODEL + hsel;
            float* dst = (row < 32) ? Ksf[row] : Vsf[row - 32];
#pragma unroll
            for (int j = 0; j < 16; ++j) dst[hsel + j] = s[j];
        }
#endif
        __syncthreads();

        // S = Q x K^T : B operand lane = key column, K-dim = head dims
        v16h kb0, kb1;
#pragma unroll
        for (int e = 0; e < 16; ++e) {
            kb0[e] = (_Float16)Ksf[nl][hi * 16 + e];
            kb1[e] = (_Float16)Ksf[16 + nl][hi * 16 + e];
        }
        v8f s0 = {}, s1 = {};
        s0 = __builtin_amdgcn_wmma_f32_16x16x32_f16(false, qa, false, kb0, (short)0, s0, false, false);
        s1 = __builtin_amdgcn_wmma_f32_16x16x32_f16(false, qa, false, kb1, (short)0, s1, false, false);

        // Online softmax; D row r of lane-group hi == query (r + 8*hi),
        // the 16 lanes of the group span the 16 key columns.
#pragma unroll
        for (int r = 0; r < 8; ++r) {
            float pm = fmaxf(s0[r], s1[r]);
            for (int msk = 1; msk < 16; msk <<= 1)
                pm = fmaxf(pm, __shfl_xor(pm, msk, 32));
            float mn   = fmaxf(m_run[r], pm);
            float resc = __expf(m_run[r] - mn);
            float p0   = __expf(s0[r] - mn);
            float p1   = __expf(s1[r] - mn);
            float ps   = p0 + p1;
            for (int msk = 1; msk < 16; msk <<= 1)
                ps += __shfl_xor(ps, msk, 32);
            l_run[r] = l_run[r] * resc + ps;
            m_run[r] = mn;
            acc0[r] *= resc;
            acc1[r] *= resc;
            Pt[wave][r + hi * 8][nl]      = (_Float16)p0; // keys k0..k0+15
            Pt[wave][r + hi * 8][16 + nl] = (_Float16)p1; // keys k0+16..k0+31
        }
        __syncthreads();

        // O += P x V : A = P (16q x 32k), B = V chunk (32k x 16d) per dim-tile
        v16h pa, vb0, vb1;
#pragma unroll
        for (int e = 0; e < 16; ++e) {
            pa[e]  = Pt[wave][nl][hi * 16 + e];
            vb0[e] = (_Float16)Vsf[hi * 16 + e][nl];
            vb1[e] = (_Float16)Vsf[hi * 16 + e][16 + nl];
        }
        acc0 = __builtin_amdgcn_wmma_f32_16x16x32_f16(false, pa, false, vb0, (short)0, acc0, false, false);
        acc1 = __builtin_amdgcn_wmma_f32_16x16x32_f16(false, pa, false, vb1, (short)0, acc1, false, false);
        __syncthreads();
    }

#pragma unroll
    for (int r = 0; r < 8; ++r) {
        float inv = 1.f / l_run[r];
        long qi = (long)b * NQ_ + q0 + r + hi * 8;
        O[qi * D_MODEL + h * HEAD_DIM + nl]      = acc0[r] * inv;
        O[qi * D_MODEL + h * HEAD_DIM + 16 + nl] = acc1[r] * inv;
    }
}

// ---------------------------------------------------------------------------
// Elementwise add
// ---------------------------------------------------------------------------
__global__ void add_kernel(const float* __restrict__ a, const float* __restrict__ b,
                           float* __restrict__ c, int n)
{
    int i = blockIdx.x * blockDim.x + threadIdx.x;
    if (i < n) c[i] = a[i] + b[i];
}

// ---------------------------------------------------------------------------
// Row LayerNorm over 256 with optional fused residual; wave per row (wave32)
// ---------------------------------------------------------------------------
__global__ void layernorm_kernel(const float* __restrict__ x, const float* __restrict__ res,
                                 const float* __restrict__ g, const float* __restrict__ bta,
                                 float* __restrict__ y, int rows)
{
    const int lane = threadIdx.x & 31;
    const int row  = blockIdx.x * (blockDim.x >> 5) + (threadIdx.x >> 5);
    if (row >= rows) return;
    const float* xp = x + (long)row * D_MODEL;
    const float* rp = res ? res + (long)row * D_MODEL : nullptr;

    float v[8];
    float s = 0.f;
#pragma unroll
    for (int j = 0; j < 8; ++j) {
        int c = lane * 8 + j;
        float t = xp[c];
        if (rp) t += rp[c];
        v[j] = t; s += t;
    }
    for (int m = 1; m < 32; m <<= 1) s += __shfl_xor(s, m, 32);
    float mu = s * (1.f / 256.f);
    float vs = 0.f;
#pragma unroll
    for (int j = 0; j < 8; ++j) { float d = v[j] - mu; vs += d * d; }
    for (int m = 1; m < 32; m <<= 1) vs += __shfl_xor(vs, m, 32);
    float rstd = rsqrtf(vs * (1.f / 256.f) + 1e-5f);

    float* yp = y + (long)row * D_MODEL;
#pragma unroll
    for (int j = 0; j < 8; ++j) {
        int c = lane * 8 + j;
        yp[c] = (v[j] - mu) * rstd * g[c] + bta[c];
    }
}

// ---------------------------------------------------------------------------
// Softmax over trailing 16 (deform attention weights), in place
// ---------------------------------------------------------------------------
__global__ void softmax16_kernel(float* __restrict__ aw, int rows)
{
    int i = blockIdx.x * blockDim.x + threadIdx.x;
    if (i >= rows) return;
    float* p = aw + (long)i * 16;
    float mx = p[0];
#pragma unroll
    for (int j = 1; j < 16; ++j) mx = fmaxf(mx, p[j]);
    float e[16], s = 0.f;
#pragma unroll
    for (int j = 0; j < 16; ++j) { e[j] = __expf(p[j] - mx); s += e[j]; }
    float inv = 1.f / s;
#pragma unroll
    for (int j = 0; j < 16; ++j) p[j] = e[j] * inv;
}

// ---------------------------------------------------------------------------
// MS-deformable attention sampling. One wave per (b, q, h); lane = head dim.
// ---------------------------------------------------------------------------
__global__ void msdeform_kernel(const float* __restrict__ ref,
                                const float* __restrict__ off,
                                const float* __restrict__ aw,
                                const float* __restrict__ value,
                                float* __restrict__ out)
{
    const int lane = threadIdx.x & 31;
    const int w = blockIdx.x * (blockDim.x >> 5) + (threadIdx.x >> 5);
    const int h  = w % N_HEADS;
    const int bq = w / N_HEADS;          // b*NQ + q
    const int b  = bq / NQ_;
    const int d  = lane;

    const int Wl_[4] = {64, 32, 16, 8};
    const int Hl_[4] = {64, 32, 16, 8};
    const int s0_[4] = {0, 4096, 5120, 5376};

    const float* refp = ref + (long)bq * 8;                       // [4][2]
    const float* offp = off + (long)bq * (N_HEADS * 32) + h * 32; // [L][P][2]
    const float* awp  = aw  + (long)bq * (N_HEADS * 16) + h * 16; // [L][P]

    float acc = 0.f;
#pragma unroll
    for (int l = 0; l < 4; ++l) {
        const float Wl = (float)Wl_[l], Hl = (float)Hl_[l];
        const float rx = refp[l * 2 + 0], ry = refp[l * 2 + 1];
#pragma unroll
        for (int p = 0; p < 4; ++p) {
            float ox = offp[(l * 4 + p) * 2 + 0];
            float oy = offp[(l * 4 + p) * 2 + 1];
            float a  = awp[l * 4 + p];
            float x  = (rx + ox / Wl) * Wl - 0.5f;
            float y  = (ry + oy / Hl) * Hl - 0.5f;
            float x0 = floorf(x), y0 = floorf(y);
            float wx1 = x - x0, wy1 = y - y0;
#pragma unroll
            for (int dy = 0; dy < 2; ++dy) {
#pragma unroll
                for (int dx = 0; dx < 2; ++dx) {
                    float xi = x0 + (float)dx, yi = y0 + (float)dy;
                    bool valid = (xi >= 0.f) && (xi <= Wl - 1.f) &&
                                 (yi >= 0.f) && (yi <= Hl - 1.f);
                    float xc = fminf(fmaxf(xi, 0.f), Wl - 1.f);
                    float yc = fminf(fmaxf(yi, 0.f), Hl - 1.f);
                    int xic = (int)xc, yic = (int)yc;
                    float wgt = (dx ? wx1 : 1.f - wx1) * (dy ? wy1 : 1.f - wy1);
                    if (valid) {
                        long s = (long)s0_[l] + (long)yic * Wl_[l] + xic;
                        acc += a * wgt *
                               value[((long)b * S_TOTAL_ + s) * D_MODEL + h * HEAD_DIM + d];
                    }
                }
            }
        }
    }
    out[(long)bq * D_MODEL + h * HEAD_DIM + d] = acc;
}

// ---------------------------------------------------------------------------
// Host orchestration
// ---------------------------------------------------------------------------
extern "C" void kernel_launch(void* const* d_in, const int* in_sizes, int n_in,
                              void* d_out, int out_size, void* d_ws, size_t ws_size,
                              hipStream_t stream)
{
    (void)in_sizes; (void)n_in; (void)out_size; (void)ws_size;

    const float* tgt       = (const float*)d_in[0];
    const float* query_pos = (const float*)d_in[1];
    const float* refpts    = (const float*)d_in[2];
    const float* src       = (const float*)d_in[3];

    float* out = (float*)d_out;
    float* ws  = (float*)d_ws;

    const int  TOK = BZ * NQ_;            // 3200
    const long NE  = (long)TOK * D_MODEL; // 819200
    const int  MV  = BZ * S_TOTAL_;       // 10880

    float* q2    = ws;
    float* qh    = q2   + NE;
    float* kh    = qh   + NE;
    float* vh    = kh   + NE;
    float* attn  = vh   + NE;
    float* t2    = attn + NE;
    float* offb  = t2   + NE;
    float* awb   = offb + NE;
    float* samp  = awb  + (long)TOK * 128;
    float* value = samp + NE;
    float* h1    = value + (long)MV * D_MODEL;

    (void)hipMemcpyAsync(out, tgt, NE * sizeof(float), hipMemcpyDeviceToDevice, stream);

    const long dd = (long)D_MODEL * D_MODEL;
    const dim3 gProj(TOK / 64, D_MODEL / 64);   // N=256
    const dim3 gAw  (TOK / 64, 128 / 64);       // N=128
    const dim3 gVal (MV / 64,  D_MODEL / 64);   // value projection
    const dim3 gFfn1(TOK / 64, D_FFN / 64);     // N=1024
    const int  lnBlocks = TOK / 4;              // 4 rows per 128-thread block

    for (int l = 0; l < NUM_LAYERS; ++l) {
        const float* sa_w_q   = (const float*)d_in[4]  + (long)l * dd;
        const float* sa_b_q   = (const float*)d_in[5]  + (long)l * D_MODEL;
        const float* sa_w_k   = (const float*)d_in[6]  + (long)l * dd;
        const float* sa_b_k   = (const float*)d_in[7]  + (long)l * D_MODEL;
        const float* sa_w_v   = (const float*)d_in[8]  + (long)l * dd;
        const float* sa_b_v   = (const float*)d_in[9]  + (long)l * D_MODEL;
        const float* sa_w_o   = (const float*)d_in[10] + (long)l * dd;
        const float* sa_b_o   = (const float*)d_in[11] + (long)l * D_MODEL;
        const float* ln2_g    = (const float*)d_in[12] + (long)l * D_MODEL;
        const float* ln2_b    = (const float*)d_in[13] + (long)l * D_MODEL;
        const float* ca_w_off = (const float*)d_in[14] + (long)l * 256 * D_MODEL;
        const float* ca_b_off = (const float*)d_in[15] + (long)l * 256;
        const float* ca_w_att = (const float*)d_in[16] + (long)l * 128 * D_MODEL;
        const float* ca_b_att = (const float*)d_in[17] + (long)l * 128;
        const float* ca_w_val = (const float*)d_in[18] + (long)l * dd;
        const float* ca_b_val = (const float*)d_in[19] + (long)l * D_MODEL;
        const float* ca_w_out = (const float*)d_in[20] + (long)l * dd;
        const float* ca_b_out = (const float*)d_in[21] + (long)l * D_MODEL;
        const float* ln1_g    = (const float*)d_in[22] + (long)l * D_MODEL;
        const float* ln1_b    = (const float*)d_in[23] + (long)l * D_MODEL;
        const float* ffn_w1   = (const float*)d_in[24] + (long)l * D_FFN * D_MODEL;
        const float* ffn_b1   = (const float*)d_in[25] + (long)l * D_FFN;
        const float* ffn_w2   = (const float*)d_in[26] + (long)l * D_MODEL * D_FFN;
        const float* ffn_b2   = (const float*)d_in[27] + (long)l * D_MODEL;
        const float* ln3_g    = (const float*)d_in[28] + (long)l * D_MODEL;
        const float* ln3_b    = (const float*)d_in[29] + (long)l * D_MODEL;

        // ---- self attention ----
        add_kernel<<<NE / 256, 256, 0, stream>>>(out, query_pos, q2, (int)NE);
        gemm_wmma_kernel<<<gProj, 128, 0, stream>>>(q2,  sa_w_q, sa_b_q, qh, TOK, D_MODEL, D_MODEL, 0);
        gemm_wmma_kernel<<<gProj, 128, 0, stream>>>(q2,  sa_w_k, sa_b_k, kh, TOK, D_MODEL, D_MODEL, 0);
        gemm_wmma_kernel<<<gProj, 128, 0, stream>>>(out, sa_w_v, sa_b_v, vh, TOK, D_MODEL, D_MODEL, 0);
        flash_attn_kernel<<<BZ * N_HEADS * (NQ_ / 16) / FA_WPB, 128, 0, stream>>>(qh, kh, vh, attn, NQ_);
        gemm_wmma_kernel<<<gProj, 128, 0, stream>>>(attn, sa_w_o, sa_b_o, t2, TOK, D_MODEL, D_MODEL, 0);
        layernorm_kernel<<<lnBlocks, 128, 0, stream>>>(out, t2, ln2_g, ln2_b, out, TOK);

        // ---- deformable cross attention ----
        add_kernel<<<NE / 256, 256, 0, stream>>>(out, query_pos, q2, (int)NE);
        gemm_wmma_kernel<<<gProj, 128, 0, stream>>>(q2, ca_w_off, ca_b_off, offb, TOK, 256, D_MODEL, 0);
        gemm_wmma_kernel<<<gAw,   128, 0, stream>>>(q2, ca_w_att, ca_b_att, awb,  TOK, 128, D_MODEL, 0);
        softmax16_kernel<<<(TOK * N_HEADS) / 256, 256, 0, stream>>>(awb, TOK * N_HEADS);
        gemm_wmma_kernel<<<gVal, 128, 0, stream>>>(src, ca_w_val, ca_b_val, value, MV, D_MODEL, D_MODEL, 0);
        msdeform_kernel<<<(TOK * N_HEADS) / 4, 128, 0, stream>>>(refpts, offb, awb, value, samp);
        gemm_wmma_kernel<<<gProj, 128, 0, stream>>>(samp, ca_w_out, ca_b_out, t2, TOK, D_MODEL, D_MODEL, 0);
        layernorm_kernel<<<lnBlocks, 128, 0, stream>>>(out, t2, ln1_g, ln1_b, out, TOK);

        // ---- FFN ----
        gemm_wmma_kernel<<<gFfn1, 128, 0, stream>>>(out, ffn_w1, ffn_b1, h1, TOK, D_FFN, D_MODEL, 1);
        gemm_wmma_kernel<<<gProj, 128, 0, stream>>>(h1,  ffn_w2, ffn_b2, t2, TOK, D_MODEL, D_FFN, 0);
        layernorm_kernel<<<lnBlocks, 128, 0, stream>>>(out, t2, ln3_g, ln3_b, out, TOK);
    }
}
